// GDNLayer_55757265436872
// MI455X (gfx1250) — compile-verified, matrix operand order checked by reading, not run on hardware
//
#include <hip/hip_runtime.h>
#include <hip/hip_bf16.h>
#include <math.h>

typedef __attribute__((ext_vector_type(16))) __bf16 v16bf;
typedef __attribute__((ext_vector_type(8)))  __bf16 v8bf;
typedef __attribute__((ext_vector_type(8)))  float  v8f;
typedef __attribute__((ext_vector_type(8)))  float  v8fv;

#define NN 50000
#define NE 800000
#define BB 2
#define WW 128
#define DD 64
#define OO 128
#define MROWS (BB * NN)   // 100000 rows for the fc GEMM (divisible by 16)
#define KIN   (2 * WW)    // 256 = concat(x, agg)

// ---------- helpers ----------

// 8 contiguous f32 -> 8 bf16 (lowers to packed v_cvt_pk_bf16_f32)
static __device__ __forceinline__ v8bf cvt8(const float* p) {
  float4 a = *(const float4*)p;
  float4 b = *(const float4*)(p + 4);
  v8fv f;
  f[0] = a.x; f[1] = a.y; f[2] = a.z; f[3] = a.w;
  f[4] = b.x; f[5] = b.y; f[6] = b.z; f[7] = b.w;
  return __builtin_convertvector(f, v8bf);
}

static __device__ __forceinline__ v16bf cat16(v8bf lo, v8bf hi) {
  v16bf r;
#pragma unroll
  for (int i = 0; i < 8; ++i) { r[i] = lo[i]; r[8 + i] = hi[i]; }
  return r;
}

static __device__ __forceinline__ float tanh_fast(float x) {
  float e = __expf(2.0f * x);
  return 1.0f - 2.0f / (e + 1.0f);
}

// ---------- kernels ----------

// Zero agg, init segment max to -inf and segment sum to 0.
__global__ __launch_bounds__(256)
void init_kernel(float* __restrict__ mx, float* __restrict__ se,
                 float* __restrict__ agg) {
  int i = blockIdx.x * 256 + threadIdx.x;
  if (i < BB * NN * WW) agg[i] = 0.0f;
  if (i < NN) { mx[i] = __int_as_float(0xFF800000); se[i] = 0.0f; }
}

// e_q = emb @ Wq^T, e_k = emb @ Wk^T via v_wmma_f32_16x16x32_bf16.
// Transposed mapping: A = weight rows (from LDS), B = embedding rows,
// D[M=out_dim][N=node] -> each lane stores 8 contiguous floats (2x b128).
__global__ __launch_bounds__(256)
void proj_kernel(const float* __restrict__ emb, const float* __restrict__ Wq,
                 const float* __restrict__ Wk, float* __restrict__ eq,
                 float* __restrict__ ek) {
  __shared__ alignas(16) __bf16 sW[2][DD * DD];  // 2 x 8 KB
  for (int i = threadIdx.x; i < DD * DD; i += 256) {
    sW[0][i] = (__bf16)Wq[i];
    sW[1][i] = (__bf16)Wk[i];
  }
  __syncthreads();

  const int lane = threadIdx.x & 31;
  const int wave = threadIdx.x >> 5;
  const int rowBase = blockIdx.x * 128 + wave * 16;
  if (rowBase >= NN) return;            // NN % 16 == 0: tiles never partial
  const int lrow = lane & 15;
  const int hi8  = (lane >> 4) * 8;     // 0 for lanes 0-15, 8 for lanes 16-31

  const int m = rowBase + lrow;         // node row owned by this lane (B column)
  const float* rp = emb + (size_t)m * DD;

  v16bf bfr[2];                         // B fragments: embedding rows, 2 k-steps
#pragma unroll
  for (int ks = 0; ks < 2; ++ks) {
    int kb = ks * 32 + hi8;
    bfr[ks] = cat16(cvt8(rp + kb), cvt8(rp + kb + 16));
  }

  float* const outp[2] = { eq, ek };
#pragma unroll
  for (int w = 0; w < 2; ++w) {
#pragma unroll
    for (int ot = 0; ot < 4; ++ot) {
      const __bf16* wrow = &sW[w][(ot * 16 + lrow) * DD];  // A: weight row
      v8f acc = {};
#pragma unroll
      for (int ks = 0; ks < 2; ++ks) {
        int kb = ks * 32 + hi8;
        v16bf af = cat16(*(const v8bf*)(wrow + kb),
                         *(const v8bf*)(wrow + kb + 16));
        acc = __builtin_amdgcn_wmma_f32_16x16x32_bf16(
            false, af, false, bfr[ks], (short)0, acc, false, false);
      }
      // D: lane holds node m, VGPR i -> out dim ot*16 + hi8 + i (contiguous)
      const int obase = ot * 16 + hi8;
      float4 r0, r1;
      r0.x = acc[0]; r0.y = acc[1]; r0.z = acc[2]; r0.w = acc[3];
      r1.x = acc[4]; r1.y = acc[5]; r1.z = acc[6]; r1.w = acc[7];
      float* op = outp[w] + (size_t)m * DD + obase;
      *(float4*)op = r0;
      *(float4*)(op + 4) = r1;
    }
  }
}

// score[e] = tanh(e_q[src] + e_k[dst]) . v ; segment max via int-trick atomics.
__global__ __launch_bounds__(256)
void score_kernel(const float* __restrict__ eq, const float* __restrict__ ek,
                  const int* __restrict__ edges, const float* __restrict__ v,
                  float* __restrict__ score, float* __restrict__ mx) {
  int e = blockIdx.x * 256 + threadIdx.x;
  if (e >= NE) return;
  int src = edges[e];
  int dst = edges[NE + e];
  const float* q = eq + (size_t)src * DD;
  const float* k = ek + (size_t)dst * DD;
  float s = 0.0f;
#pragma unroll
  for (int i = 0; i < DD; i += 4) {
    float4 qv = *(const float4*)(q + i);
    float4 kv = *(const float4*)(k + i);
    float4 vv = *(const float4*)(v + i);
    s += tanh_fast(qv.x + kv.x) * vv.x + tanh_fast(qv.y + kv.y) * vv.y +
         tanh_fast(qv.z + kv.z) * vv.z + tanh_fast(qv.w + kv.w) * vv.w;
  }
  score[e] = s;
  // float atomic max (mx init'd to -inf): signed max for >=0, unsigned min for <0
  if (s >= 0.0f) atomicMax((int*)(mx + dst), __float_as_int(s));
  else           atomicMin((unsigned int*)(mx + dst),
                           (unsigned int)__float_as_int(s));
}

// exp(score - max[dst]) in place + segment sum.
__global__ __launch_bounds__(256)
void expsum_kernel(const int* __restrict__ edges, float* __restrict__ score,
                   const float* __restrict__ mx, float* __restrict__ se) {
  int e = blockIdx.x * 256 + threadIdx.x;
  if (e >= NE) return;
  int dst = edges[NE + e];
  float ex = __expf(score[e] - mx[dst]);
  score[e] = ex;
  atomicAdd(&se[dst], ex);
}

// One wave per (edge, batch): lane handles 4 floats -> 128-wide coalesced
// gather of x[b,src,:] and 4 f32 atomic adds into agg[b,dst,:] (L2-resident).
__global__ __launch_bounds__(256)
void agg_kernel(const float* __restrict__ x, const int* __restrict__ edges,
                const float* __restrict__ expsc, const float* __restrict__ se,
                float* __restrict__ agg) {
  int gid  = blockIdx.x * 256 + threadIdx.x;
  int wid  = gid >> 5;
  int lane = gid & 31;
  if (wid >= NE * BB) return;
  int e = wid >> 1;
  int b = wid & 1;
  int src = edges[e];
  int dst = edges[NE + e];
  float attn = expsc[e] / (se[dst] + 1e-8f);
  const float* xr = x   + ((size_t)b * NN + src) * WW + lane * 4;
  float*       ar = agg + ((size_t)b * NN + dst) * WW + lane * 4;
  float4 xv = *(const float4*)xr;
  atomicAdd(ar + 0, attn * xv.x);
  atomicAdd(ar + 1, attn * xv.y);
  atomicAdd(ar + 2, attn * xv.z);
  atomicAdd(ar + 3, attn * xv.w);
}

// out = relu(concat(x, agg) @ fcW^T + fcb): M=100000, K=256, N=128.
// Transposed mapping: A = fcW rows from LDS (row-major, no transpose needed),
// B = activation rows; lane stores 8 contiguous outputs -> 2x global_store_b128.
__global__ __launch_bounds__(256)
void fc_kernel(const float* __restrict__ x, const float* __restrict__ agg,
               const float* __restrict__ fcW, const float* __restrict__ fcb,
               float* __restrict__ out) {
  __shared__ alignas(16) __bf16 sW[OO * KIN];  // 128*256*2B = 64 KB
  for (int i = threadIdx.x; i < OO * KIN; i += 256) sW[i] = (__bf16)fcW[i];
  __syncthreads();

  const int lane = threadIdx.x & 31;
  const int wave = threadIdx.x >> 5;
  const int rowBase = blockIdx.x * 128 + wave * 16;
  if (rowBase >= MROWS) return;         // MROWS % 16 == 0: tiles never partial
  const int lrow = lane & 15;
  const int hi8  = (lane >> 4) * 8;

  const int m = rowBase + lrow;         // flat row = b*NN + n (x/agg/out agree)
  const float* xr = x   + (size_t)m * WW;
  const float* ar = agg + (size_t)m * WW;

  v16bf bfr[8];                         // B fragments: activation rows
#pragma unroll
  for (int ks = 0; ks < 8; ++ks) {
    int kb = ks * 32 + hi8;
    int k0 = kb, k1 = kb + 16;          // 8-wide chunks never straddle the split
    const float* p0 = (k0 < WW) ? (xr + k0) : (ar + (k0 - WW));
    const float* p1 = (k1 < WW) ? (xr + k1) : (ar + (k1 - WW));
    bfr[ks] = cat16(cvt8(p0), cvt8(p1));
  }

#pragma unroll
  for (int ot = 0; ot < 8; ++ot) {
    const __bf16* wrow = &sW[(ot * 16 + lrow) * KIN];  // A: weight row
    v8f acc = {};
#pragma unroll
    for (int ks = 0; ks < 8; ++ks) {
      int kb = ks * 32 + hi8;
      v16bf af = cat16(*(const v8bf*)(wrow + kb),
                       *(const v8bf*)(wrow + kb + 16));
      acc = __builtin_amdgcn_wmma_f32_16x16x32_bf16(
          false, af, false, bfr[ks], (short)0, acc, false, false);
    }
    // D: lane holds row m; VGPR i -> output feature obase + i (contiguous)
    const int obase = ot * 16 + hi8;
    float4 b0 = *(const float4*)(fcb + obase);
    float4 b1 = *(const float4*)(fcb + obase + 4);
    float4 r0, r1;
    r0.x = fmaxf(acc[0] + b0.x, 0.0f);
    r0.y = fmaxf(acc[1] + b0.y, 0.0f);
    r0.z = fmaxf(acc[2] + b0.z, 0.0f);
    r0.w = fmaxf(acc[3] + b0.w, 0.0f);
    r1.x = fmaxf(acc[4] + b1.x, 0.0f);
    r1.y = fmaxf(acc[5] + b1.y, 0.0f);
    r1.z = fmaxf(acc[6] + b1.z, 0.0f);
    r1.w = fmaxf(acc[7] + b1.w, 0.0f);
    float* op = out + (size_t)m * OO + obase;
    *(float4*)op = r0;
    *(float4*)(op + 4) = r1;
  }
}

// ---------- launch ----------

extern "C" void kernel_launch(void* const* d_in, const int* in_sizes, int n_in,
                              void* d_out, int out_size, void* d_ws, size_t ws_size,
                              hipStream_t stream) {
  const float* x     = (const float*)d_in[0];
  const float* emb   = (const float*)d_in[1];
  const int*   edges = (const int*)d_in[2];
  const float* Wq    = (const float*)d_in[3];
  const float* Wk    = (const float*)d_in[4];
  const float* v     = (const float*)d_in[5];
  const float* fcW   = (const float*)d_in[6];
  const float* fcb   = (const float*)d_in[7];
  float* out = (float*)d_out;

  float* ws  = (float*)d_ws;
  float* eq  = ws;                       // NN*DD
  float* ek  = eq + (size_t)NN * DD;     // NN*DD
  float* sc  = ek + (size_t)NN * DD;     // NE (score -> exp(score-max))
  float* mx  = sc + NE;                  // NN
  float* se  = mx + NN;                  // NN
  float* agg = se + NN;                  // BB*NN*WW

  init_kernel  <<<(BB * NN * WW + 255) / 256, 256, 0, stream>>>(mx, se, agg);
  proj_kernel  <<<(NN + 127) / 128,          256, 0, stream>>>(emb, Wq, Wk, eq, ek);
  score_kernel <<<(NE + 255) / 256,          256, 0, stream>>>(eq, ek, edges, v, sc, mx);
  expsum_kernel<<<(NE + 255) / 256,          256, 0, stream>>>(edges, sc, mx, se);
  agg_kernel   <<<(int)(((size_t)NE * BB * 32 + 255) / 256), 256, 0, stream>>>(x, edges, sc, se, agg);
  fc_kernel    <<<(MROWS + 127) / 128,       256, 0, stream>>>(x, agg, fcW, fcb, out);
}
